// RawKVInjector_20615843021108
// MI455X (gfx1250) — compile-verified, matrix operand order checked by reading, not run on hardware
//
#include <hip/hip_runtime.h>

// ---------------------------------------------------------------------------
// MI455X (gfx1250) decoder layer with virtual-KV injection.
// All matmuls -> v_wmma_f32_16x16x32_bf16 (bf16 in, f32 accumulate).
// GEMM operand staging -> Tensor Data Mover (tensor_load_to_lds) with LDS
// padding generated by the TDM pad unit, double buffered, s_wait_tensorcnt.
// ---------------------------------------------------------------------------

typedef __attribute__((ext_vector_type(16))) __bf16 v16bf;
typedef __attribute__((ext_vector_type(8)))  __bf16 v8bf;
typedef __attribute__((ext_vector_type(8)))  float  v8f;
typedef __attribute__((ext_vector_type(4)))  unsigned int u32x4;
typedef __attribute__((ext_vector_type(8)))  int i32x8;
typedef __attribute__((ext_vector_type(4)))  int i32x4;

static_assert(sizeof(__bf16) == 2, "bf16 must be 2 bytes");

#if defined(__has_builtin)
#if __has_builtin(__builtin_amdgcn_tensor_load_to_lds) && \
    __has_builtin(__builtin_amdgcn_s_wait_tensorcnt)
#define USE_TDM 1
#endif
#endif
#ifndef USE_TDM
#define USE_TDM 0
#endif

// ---- problem constants (match reference) ----------------------------------
constexpr int B_   = 2;
constexpr int T_   = 2048;
constexpr int D_   = 2048;
constexpr int H_   = 16;
constexpr int HKV_ = 8;
constexpr int HD_  = 128;
constexpr int R_   = 128;
constexpr int DFF_ = 6144;
constexpr int TK_  = R_ + T_;        // 2176
constexpr int BT_  = B_ * T_;        // 4096
constexpr float SCALING_ = 0.08838834764831845f;   // HD^-0.5

// ---- workspace layout (bytes, all 256-aligned) ----------------------------
// Weights are stored bf16 *transposed* (N-major, row length K) so every GEMM
// operand tile is a contiguous-row 2D tile (TDM friendly).
constexpr size_t OFF_WQ  = 0;                         // (2048 x 2048) bf16
constexpr size_t OFF_WK  = OFF_WQ + 8388608ull;       // (1024 x 2048) bf16
constexpr size_t OFF_WV  = OFF_WK + 4194304ull;
constexpr size_t OFF_WO  = OFF_WV + 4194304ull;       // (2048 x 2048) bf16
constexpr size_t OFF_WG  = OFF_WO + 8388608ull;       // (6144 x 2048) bf16
constexpr size_t OFF_WU  = OFF_WG + 25165824ull;
constexpr size_t OFF_WD  = OFF_WU + 25165824ull;      // (2048 x 6144) bf16
constexpr size_t OFF_HN  = OFF_WD + 25165824ull;      // normed hidden bf16
constexpr size_t OFF_HID = OFF_HN + 16777216ull;      // hidden f32 after attn
constexpr size_t SCR     = OFF_HID + 33554432ull;     // phase-reused scratch
// attention phase
constexpr size_t OFF_QRAW = SCR;                       // (BT,2048) bf16
constexpr size_t OFF_KRAW = OFF_QRAW + 16777216ull;    // (BT,1024) bf16
constexpr size_t OFF_VRAW = OFF_KRAW + 8388608ull;
constexpr size_t OFF_QALL = OFF_VRAW + 8388608ull;     // (B,H,T,HD) bf16
constexpr size_t OFF_KALL = OFF_QALL + 16777216ull;    // (B,HKV,TK,HD) bf16
constexpr size_t OFF_VALL = OFF_KALL + 8912896ull;
constexpr size_t OFF_ATTN = OFF_VALL + 8912896ull;     // (BT,2048) bf16
// MLP phase (overlaps attention scratch; used only after hidden is computed)
constexpr size_t OFF_G = SCR;                          // (BT,DFF) bf16
constexpr size_t OFF_U = OFF_G + 50331648ull;
constexpr size_t OFF_M = OFF_U + 50331648ull;

// ---------------------------------------------------------------------------
// WMMA helpers
// ---------------------------------------------------------------------------
__device__ __forceinline__ v8f wmma_bf16(v16bf a, v16bf b, v8f c) {
  return __builtin_amdgcn_wmma_f32_16x16x32_bf16(
      false, a, false, b, (short)0, c, false, false);
}

__device__ __forceinline__ v8f vzero8() {
  v8f z;
#pragma unroll
  for (int i = 0; i < 8; ++i) z[i] = 0.0f;
  return z;
}

// 16x32 bf16 fragment load (A layout; also B layout when operand stored
// N-major).  Per ISA: lanes 0-15 row m, K 0..7 / 16..23; lanes 16-31 row m,
// K 8..15 / 24..31 -> two contiguous 16B chunks per lane.
__device__ __forceinline__ v16bf frag_load(const __bf16* base, int ld, int lane) {
  const int m  = lane & 15;
  const int kb = (lane >> 4) << 3;
  const __bf16* p = base + m * ld;
  v8bf lo = *(const v8bf*)(p + kb);
  v8bf hi = *(const v8bf*)(p + kb + 16);
  v16bf r;
#pragma unroll
  for (int i = 0; i < 8; ++i) { r[i] = lo[i]; r[i + 8] = hi[i]; }
  return r;
}

#if USE_TDM
// ---------------------------------------------------------------------------
// TDM: load a 2D bf16 tile (tileRows x 32 elements, row stride = rowLen elems)
// into LDS with 16 data DWORDs + 4 pad DWORDs per row => LDS row pitch of 40
// bf16 elements, matching frag_load's ld=40.
//   D# group0: count=1 | lds_addr | global_addr | type=2
//   D# group1: data_size=2B, pad_enable, pad_interval=16DW(code 3),
//              pad_amount=4DW(code 3), dims/strides, tile_dim0=32
// 6-arg builtin (clang-23 / therock headers): g0, g1, g2, g3, g4, cpol.
// ---------------------------------------------------------------------------
__device__ __forceinline__ void tdm_load_tile32(unsigned ldsOff,
                                                const __bf16* gptr,
                                                int tileRows, int rowLen,
                                                int totRows) {
  const unsigned long long ga = (unsigned long long)(size_t)gptr;
  u32x4 g0;
  g0[0] = 1u;                                        // count=1, user mode
  g0[1] = ldsOff;                                    // lds_addr (bytes)
  g0[2] = (unsigned)ga;                              // global_addr[31:0]
  g0[3] = (unsigned)((ga >> 32) & 0x1FFFFFFull) | (2u << 30);  // [56:32]|type=2
  i32x8 g1;
  g1[0] = (int)((1u << 16) |   // data_size = 2 bytes
                (1u << 20) |   // pad_enable
                (3u << 22) |   // pad_interval: 16 DWORDs of data
                (3u << 25));   // pad_amount:   4 DWORDs of pad
  g1[1] = (int)(((unsigned)rowLen & 0xFFFFu) << 16);            // dim0[15:0]
  g1[2] = (int)((((unsigned)rowLen >> 16) & 0xFFFFu) |
                (((unsigned)totRows & 0xFFFFu) << 16));          // dim0 hi|dim1 lo
  g1[3] = (int)((((unsigned)totRows >> 16) & 0xFFFFu) |
                (32u << 16));                                    // dim1 hi|tile0=32
  g1[4] = tileRows & 0xFFFF;                                     // tile_dim1
  g1[5] = rowLen;                                                // dim0_stride lo
  g1[6] = 0;
  g1[7] = 0;
  i32x4 z4;
  z4[0] = 0; z4[1] = 0; z4[2] = 0; z4[3] = 0;
  i32x8 z8;
#pragma unroll
  for (int i = 0; i < 8; ++i) z8[i] = 0;
  __builtin_amdgcn_tensor_load_to_lds(g0, g1, z4, z4, z8, 0);
}
#endif

// ---------------------------------------------------------------------------
// f32 -> bf16 transpose convert for weights: in (K x N) f32 -> out (N x K) bf16
// ---------------------------------------------------------------------------
__global__ __launch_bounds__(256) void transpose_bf16_kern(
    const float* __restrict__ in, __bf16* __restrict__ out, int K, int N) {
  __shared__ __bf16 tile[32][33];
  const int tx = threadIdx.x & 31;
  const int ty = threadIdx.x >> 5;           // 0..7
  const int k0 = blockIdx.y * 32;
  const int n0 = blockIdx.x * 32;
#pragma unroll
  for (int i = 0; i < 32; i += 8)
    tile[ty + i][tx] = (__bf16)in[(size_t)(k0 + ty + i) * N + n0 + tx];
  __syncthreads();
#pragma unroll
  for (int i = 0; i < 32; i += 8)
    out[(size_t)(n0 + ty + i) * K + k0 + tx] = tile[tx][ty + i];
}

// ---------------------------------------------------------------------------
// RMSNorm over D=2048 (f32 in, bf16 out), one block per row
// ---------------------------------------------------------------------------
__global__ __launch_bounds__(256) void rmsnorm_kern(const float* __restrict__ x,
                                                    const float* __restrict__ w,
                                                    __bf16* __restrict__ out) {
  __shared__ float red[8];
  const int row = blockIdx.x;
  const float* xr = x + (size_t)row * D_;
  float ss = 0.f;
  for (int i = threadIdx.x; i < D_; i += 256) { float v = xr[i]; ss += v * v; }
#pragma unroll
  for (int off = 16; off > 0; off >>= 1) ss += __shfl_xor(ss, off, 32);
  if ((threadIdx.x & 31) == 0) red[threadIdx.x >> 5] = ss;
  __syncthreads();
  float tot = 0.f;
#pragma unroll
  for (int i = 0; i < 8; ++i) tot += red[i];
  const float inv = rsqrtf(tot * (1.0f / D_) + 1e-6f);
  __bf16* o = out + (size_t)row * D_;
  for (int i = threadIdx.x; i < D_; i += 256) o[i] = (__bf16)(xr[i] * inv * w[i]);
}

// ---------------------------------------------------------------------------
// bf16 WMMA GEMM: C(MxN) = A(MxK) * Bt(NxK)^T  [+ res], f32 or bf16 out.
// 256 thr / 8 waves, tile 128x64, BK=32.  Operand staging via TDM double
// buffering (wave 0 issues DMA; s_wait_tensorcnt; barrier publishes tile).
// ---------------------------------------------------------------------------
__global__ __launch_bounds__(256) void gemm_bf16_kern(
    const __bf16* __restrict__ A, const __bf16* __restrict__ Bt,
    int M, int N, int K,
    float* outF, __bf16* outB, const float* __restrict__ res) {
  __shared__ __bf16 As[2][128 * 40];   // [row][k] pitch 40
  __shared__ __bf16 Bs[2][64 * 40];    // [col][k] pitch 40

  const int tid  = threadIdx.x;
  const int lane = tid & 31;
  const int wave = tid >> 5;
  const int wm   = wave & 3;
  const int wn   = wave >> 2;
  const int row0 = blockIdx.y * 128;
  const int col0 = blockIdx.x * 64;
  const int nSteps = K >> 5;

  v8f acc[2][2];
#pragma unroll
  for (int i = 0; i < 2; ++i)
#pragma unroll
    for (int j = 0; j < 2; ++j) acc[i][j] = vzero8();

#if USE_TDM
  const unsigned ldsA0 = (unsigned)(size_t)&As[0][0];
  const unsigned ldsA1 = (unsigned)(size_t)&As[1][0];
  const unsigned ldsB0 = (unsigned)(size_t)&Bs[0][0];
  const unsigned ldsB1 = (unsigned)(size_t)&Bs[1][0];
  const __bf16* aBase = A  + (size_t)row0 * K;
  const __bf16* bBase = Bt + (size_t)col0 * K;
  if (wave == 0) {
    tdm_load_tile32(ldsA0, aBase, 128, K, M);
    tdm_load_tile32(ldsB0, bBase, 64,  K, N);
  }
  for (int i = 0; i < nSteps; ++i) {
    const int cur = i & 1;
    if (wave == 0) {
      if (i + 1 < nSteps) {
        const int k1 = (i + 1) << 5;
        tdm_load_tile32(cur ? ldsA0 : ldsA1, aBase + k1, 128, K, M);
        tdm_load_tile32(cur ? ldsB0 : ldsB1, bBase + k1, 64,  K, N);
        __builtin_amdgcn_s_wait_tensorcnt(2);   // previous pair landed
      } else {
        __builtin_amdgcn_s_wait_tensorcnt(0);
      }
    }
    __syncthreads();
#pragma unroll
    for (int ii = 0; ii < 2; ++ii) {
      v16bf a = frag_load(&As[cur][(wm * 32 + ii * 16) * 40], 40, lane);
#pragma unroll
      for (int jj = 0; jj < 2; ++jj) {
        v16bf b = frag_load(&Bs[cur][(wn * 32 + jj * 16) * 40], 40, lane);
        acc[ii][jj] = wmma_bf16(a, b, acc[ii][jj]);
      }
    }
    __syncthreads();
  }
#else
  for (int s = 0; s < nSteps; ++s) {
    const int k0 = s << 5;
    const int cur = s & 1;
    __syncthreads();
    {  // A tile 128x32: 16 bf16/thread, contiguous both sides
      const int r = tid >> 1;
      const int c = (tid & 1) << 4;
      const __bf16* g = A + (size_t)(row0 + r) * K + k0 + c;
      *(v8bf*)&As[cur][r * 40 + c]     = *(const v8bf*)g;
      *(v8bf*)&As[cur][r * 40 + c + 8] = *(const v8bf*)(g + 8);
    }
    {  // B tile 64x32 from transposed weights: contiguous both sides
      const int n  = tid >> 2;
      const int kk = (tid & 3) << 3;
      const __bf16* g = Bt + (size_t)(col0 + n) * K + k0 + kk;
      *(v8bf*)&Bs[cur][n * 40 + kk] = *(const v8bf*)g;
    }
    __syncthreads();
#pragma unroll
    for (int ii = 0; ii < 2; ++ii) {
      v16bf a = frag_load(&As[cur][(wm * 32 + ii * 16) * 40], 40, lane);
#pragma unroll
      for (int jj = 0; jj < 2; ++jj) {
        v16bf b = frag_load(&Bs[cur][(wn * 32 + jj * 16) * 40], 40, lane);
        acc[ii][jj] = wmma_bf16(a, b, acc[ii][jj]);
      }
    }
  }
#endif

  // epilogue: D fragment -> row = r + 8*(lane>=16), col = lane&15
  const int rlo = (lane >> 4) << 3;
  const int cc  = lane & 15;
#pragma unroll
  for (int i = 0; i < 2; ++i)
#pragma unroll
    for (int j = 0; j < 2; ++j) {
      const int col = col0 + wn * 32 + j * 16 + cc;
#pragma unroll
      for (int r = 0; r < 8; ++r) {
        const int row = row0 + wm * 32 + i * 16 + rlo + r;
        float v = acc[i][j][r];
        if (res)  v += res[(size_t)row * N + col];
        if (outF) outF[(size_t)row * N + col] = v;
        else      outB[(size_t)row * N + col] = (__bf16)v;
      }
    }
}

// ---------------------------------------------------------------------------
// Per-head RMSNorm + RoPE (+layout move).  One wave per (b, head, t).
// mode 0: q-norm+rope, 1: k-norm+rope, 2: v passthrough.
// ---------------------------------------------------------------------------
__global__ __launch_bounds__(256) void head_norm_rope_kern(
    const __bf16* __restrict__ raw, const float* __restrict__ nw,
    const float* __restrict__ cosT, const float* __restrict__ sinT,
    __bf16* __restrict__ outAll, int nHeads, int tkStride, int tOff, int mode) {
  const int wave = threadIdx.x >> 5;
  const int lane = threadIdx.x & 31;
  const int row  = blockIdx.x * 8 + wave;
  const int t  = row % T_;
  const int bh = row / T_;
  const int h  = bh % nHeads;
  const int b  = bh / nHeads;
  const int d0 = lane * 4;

  const __bf16* in = raw + ((size_t)(b * T_ + t) * nHeads + h) * HD_ + d0;
  float x[4];
#pragma unroll
  for (int i = 0; i < 4; ++i) x[i] = (float)in[i];

  __bf16* o = outAll + ((size_t)(b * nHeads + h) * tkStride + tOff + t) * HD_ + d0;

  if (mode == 2) {
#pragma unroll
    for (int i = 0; i < 4; ++i) o[i] = (__bf16)x[i];
    return;
  }

  float ss = x[0]*x[0] + x[1]*x[1] + x[2]*x[2] + x[3]*x[3];
#pragma unroll
  for (int off = 16; off > 0; off >>= 1) ss += __shfl_xor(ss, off, 32);
  const float inv = rsqrtf(ss * (1.0f / HD_) + 1e-6f);

  const int pd0 = (d0 + 64) & 127;
  const __bf16* inp = raw + ((size_t)(b * T_ + t) * nHeads + h) * HD_ + pd0;
  float xp[4];
#pragma unroll
  for (int i = 0; i < 4; ++i) xp[i] = (float)inp[i];

#pragma unroll
  for (int i = 0; i < 4; ++i) {
    const int d  = d0 + i;
    const int pd = (d + 64) & 127;
    const float xn  = x[i]  * inv * nw[d];
    const float pn  = xp[i] * inv * nw[pd];
    const float rot = (d < 64) ? -pn : pn;
    o[i] = (__bf16)(xn * cosT[(size_t)t * HD_ + d] + rot * sinT[(size_t)t * HD_ + d]);
  }
}

// ---------------------------------------------------------------------------
// virtual-KV injection: k_all/v_all[0..R) = bf16(extra * 0.5*sigmoid(logit))
// ---------------------------------------------------------------------------
__global__ void inject_extras_kern(const float* __restrict__ ke,
                                   const float* __restrict__ ve,
                                   const float* __restrict__ alpha_logit,
                                   __bf16* __restrict__ kAll,
                                   __bf16* __restrict__ vAll) {
  const size_t i = (size_t)blockIdx.x * 256 + threadIdx.x;
  const float eff = 0.5f / (1.0f + __expf(-alpha_logit[0]));
  const int d = (int)(i & (HD_ - 1));
  const size_t rest = i >> 7;
  const int r = (int)(rest & (R_ - 1));
  const size_t bh = rest >> 7;
  const size_t o = (bh * TK_ + r) * HD_ + d;
  kAll[o] = (__bf16)(ke[i] * eff);
  vAll[o] = (__bf16)(ve[i] * eff);
}

// ---------------------------------------------------------------------------
// WMMA flash attention: 128 always-visible virtual tokens + causal rest.
// block = (128-query tile, head, batch); 8 waves; wave owns 16 query rows.
// ---------------------------------------------------------------------------
__global__ __launch_bounds__(256) void attn_kern(
    const __bf16* __restrict__ q, const __bf16* __restrict__ kAll,
    const __bf16* __restrict__ vAll, __bf16* __restrict__ out) {
  __shared__ __bf16 Ks[64 * 136];        // [key][hd]      ld=136
  __shared__ __bf16 Vst[128 * 72];       // [hd][key]      ld=72
  __shared__ __bf16 Ps[8 * 16 * 72];     // per-wave [row][key] ld=72

  const int tid  = threadIdx.x;
  const int lane = tid & 31;
  const int wave = tid >> 5;
  const int q0   = blockIdx.x * 128;
  const int h    = blockIdx.y;
  const int b    = blockIdx.z;
  const int hkv  = h >> 1;                      // G = 2

  const __bf16* qBase = q + ((size_t)(b * H_ + h) * T_ + q0 + wave * 16) * HD_;
  v16bf aq[4];
#pragma unroll
  for (int f = 0; f < 4; ++f) aq[f] = frag_load(qBase + f * 32, HD_, lane);

  v8f accO[8];
#pragma unroll
  for (int t = 0; t < 8; ++t) accO[t] = vzero8();
  float mrun[8], lrun[8];
#pragma unroll
  for (int r = 0; r < 8; ++r) { mrun[r] = -1e30f; lrun[r] = 0.f; }

  const int rowBase = q0 + wave * 16 + ((lane >> 4) << 3);
  const size_t kvBase = (size_t)(b * HKV_ + hkv) * TK_;
  const int nChunks = (R_ + q0 + 128) / 64;

  for (int c = 0; c < nChunks; ++c) {
    const int kc = c * 64;
    __syncthreads();
    {  // K chunk -> LDS key-major
      const int idx = tid * 32;
      const int key = idx >> 7;
      const int d   = idx & 127;
      const __bf16* g = kAll + (kvBase + kc + key) * HD_ + d;
#pragma unroll
      for (int i = 0; i < 4; ++i)
        *(v8bf*)&Ks[key * 136 + d + i * 8] = *(const v8bf*)(g + i * 8);
    }
    {  // V chunk -> LDS transposed [hd][key]
      const int idx = tid * 32;
      const int key = idx >> 7;
      const int d   = idx & 127;
      const __bf16* g = vAll + (kvBase + kc + key) * HD_ + d;
#pragma unroll
      for (int i = 0; i < 32; ++i) Vst[(d + i) * 72 + key] = g[i];
    }
    __syncthreads();

    v8f s[4];
#pragma unroll
    for (int j = 0; j < 4; ++j) {
      s[j] = vzero8();
#pragma unroll
      for (int f = 0; f < 4; ++f) {
        v16bf bk = frag_load(&Ks[(j * 16) * 136 + f * 32], 136, lane);
        s[j] = wmma_bf16(aq[f], bk, s[j]);
      }
    }
#pragma unroll
    for (int j = 0; j < 4; ++j) {
      const int key = kc + j * 16 + (lane & 15);
#pragma unroll
      for (int r = 0; r < 8; ++r) {
        float sv = s[j][r] * SCALING_;
        if (key >= R_ && (key - R_) > (rowBase + r)) sv = -1e9f;
        s[j][r] = sv;
      }
    }
    float mnew[8], rs[8], sf[8];
#pragma unroll
    for (int r = 0; r < 8; ++r) {
      float t0 = fmaxf(fmaxf(s[0][r], s[1][r]), fmaxf(s[2][r], s[3][r]));
#pragma unroll
      for (int off = 1; off < 16; off <<= 1) t0 = fmaxf(t0, __shfl_xor(t0, off, 32));
      mnew[r] = fmaxf(mrun[r], t0);
      rs[r] = 0.f;
    }
#pragma unroll
    for (int j = 0; j < 4; ++j)
#pragma unroll
      for (int r = 0; r < 8; ++r) {
        const float p = __expf(s[j][r] - mnew[r]);
        s[j][r] = p;
        rs[r] += p;
      }
#pragma unroll
    for (int r = 0; r < 8; ++r) {
#pragma unroll
      for (int off = 1; off < 16; off <<= 1) rs[r] += __shfl_xor(rs[r], off, 32);
      sf[r] = __expf(mrun[r] - mnew[r]);
      lrun[r] = lrun[r] * sf[r] + rs[r];
      mrun[r] = mnew[r];
    }
#pragma unroll
    for (int t = 0; t < 8; ++t)
#pragma unroll
      for (int r = 0; r < 8; ++r) accO[t][r] *= sf[r];

    __bf16* pw = &Ps[wave * 16 * 72];
    const int prow = (lane >> 4) << 3;
#pragma unroll
    for (int j = 0; j < 4; ++j)
#pragma unroll
      for (int r = 0; r < 8; ++r)
        pw[(prow + r) * 72 + j * 16 + (lane & 15)] = (__bf16)s[j][r];
    v16bf ap[2];
#pragma unroll
    for (int g2 = 0; g2 < 2; ++g2) ap[g2] = frag_load(pw + g2 * 32, 72, lane);

#pragma unroll
    for (int t = 0; t < 8; ++t)
#pragma unroll
      for (int g2 = 0; g2 < 2; ++g2) {
        v16bf bv = frag_load(&Vst[(t * 16) * 72 + g2 * 32], 72, lane);
        accO[t] = wmma_bf16(ap[g2], bv, accO[t]);
      }
  }

#pragma unroll
  for (int r = 0; r < 8; ++r) {
    const float inv = 1.0f / lrun[r];
    const int qi = rowBase + r;
    __bf16* o = out + (size_t)(b * T_ + qi) * (H_ * HD_) + h * HD_;
#pragma unroll
    for (int t = 0; t < 8; ++t)
      o[t * 16 + (lane & 15)] = (__bf16)(accO[t][r] * inv);
  }
}

// ---------------------------------------------------------------------------
// SwiGLU combine: m = silu(g) * u
// ---------------------------------------------------------------------------
__global__ void swiglu_kern(const __bf16* __restrict__ g,
                            const __bf16* __restrict__ u,
                            __bf16* __restrict__ m, size_t n) {
  size_t i = (size_t)blockIdx.x * blockDim.x + threadIdx.x;
  size_t stride = (size_t)gridDim.x * blockDim.x;
  for (; i < n; i += stride) {
    const float gv = (float)g[i];
    const float uv = (float)u[i];
    m[i] = (__bf16)((gv / (1.0f + __expf(-gv))) * uv);
  }
}

// ---------------------------------------------------------------------------
// host orchestration
// ---------------------------------------------------------------------------
extern "C" void kernel_launch(void* const* d_in, const int* in_sizes, int n_in,
                              void* d_out, int out_size, void* d_ws, size_t ws_size,
                              hipStream_t stream) {
  (void)in_sizes; (void)n_in; (void)out_size; (void)ws_size;
  const float* hidden_states = (const float*)d_in[0];
  const float* key_extra     = (const float*)d_in[1];
  const float* value_extra   = (const float*)d_in[2];
  const float* cosT          = (const float*)d_in[3];
  const float* sinT          = (const float*)d_in[4];
  const float* alpha_logit   = (const float*)d_in[6];
  const float* Wq = (const float*)d_in[7];
  const float* Wk = (const float*)d_in[8];
  const float* Wv = (const float*)d_in[9];
  const float* Wo = (const float*)d_in[10];
  const float* q_norm_w = (const float*)d_in[11];
  const float* k_norm_w = (const float*)d_in[12];
  const float* ln1_w = (const float*)d_in[13];
  const float* ln2_w = (const float*)d_in[14];
  const float* Wg = (const float*)d_in[15];
  const float* Wu = (const float*)d_in[16];
  const float* Wd = (const float*)d_in[17];

  char* ws = (char*)d_ws;
  __bf16* wq_b = (__bf16*)(ws + OFF_WQ);
  __bf16* wk_b = (__bf16*)(ws + OFF_WK);
  __bf16* wv_b = (__bf16*)(ws + OFF_WV);
  __bf16* wo_b = (__bf16*)(ws + OFF_WO);
  __bf16* wg_b = (__bf16*)(ws + OFF_WG);
  __bf16* wu_b = (__bf16*)(ws + OFF_WU);
  __bf16* wd_b = (__bf16*)(ws + OFF_WD);
  __bf16* hn_b = (__bf16*)(ws + OFF_HN);
  float*  hid  = (float*)(ws + OFF_HID);
  __bf16* qraw = (__bf16*)(ws + OFF_QRAW);
  __bf16* kraw = (__bf16*)(ws + OFF_KRAW);
  __bf16* vraw = (__bf16*)(ws + OFF_VRAW);
  __bf16* qall = (__bf16*)(ws + OFF_QALL);
  __bf16* kall = (__bf16*)(ws + OFF_KALL);
  __bf16* vall = (__bf16*)(ws + OFF_VALL);
  __bf16* attn = (__bf16*)(ws + OFF_ATTN);
  __bf16* g_b  = (__bf16*)(ws + OFF_G);
  __bf16* u_b  = (__bf16*)(ws + OFF_U);
  __bf16* m_b  = (__bf16*)(ws + OFF_M);

  // 1) weights -> bf16 transposed (N-major, row length K)
  transpose_bf16_kern<<<dim3(2048/32, 2048/32), 256, 0, stream>>>(Wq, wq_b, 2048, 2048);
  transpose_bf16_kern<<<dim3(1024/32, 2048/32), 256, 0, stream>>>(Wk, wk_b, 2048, 1024);
  transpose_bf16_kern<<<dim3(1024/32, 2048/32), 256, 0, stream>>>(Wv, wv_b, 2048, 1024);
  transpose_bf16_kern<<<dim3(2048/32, 2048/32), 256, 0, stream>>>(Wo, wo_b, 2048, 2048);
  transpose_bf16_kern<<<dim3(6144/32, 2048/32), 256, 0, stream>>>(Wg, wg_b, 2048, 6144);
  transpose_bf16_kern<<<dim3(6144/32, 2048/32), 256, 0, stream>>>(Wu, wu_b, 2048, 6144);
  transpose_bf16_kern<<<dim3(2048/32, 6144/32), 256, 0, stream>>>(Wd, wd_b, 6144, 2048);

  // 2) ln1
  rmsnorm_kern<<<BT_, 256, 0, stream>>>(hidden_states, ln1_w, hn_b);

  // 3) QKV projections
  gemm_bf16_kern<<<dim3(2048/64, BT_/128), 256, 0, stream>>>(
      hn_b, wq_b, BT_, 2048, 2048, nullptr, qraw, nullptr);
  gemm_bf16_kern<<<dim3(1024/64, BT_/128), 256, 0, stream>>>(
      hn_b, wk_b, BT_, 1024, 2048, nullptr, kraw, nullptr);
  gemm_bf16_kern<<<dim3(1024/64, BT_/128), 256, 0, stream>>>(
      hn_b, wv_b, BT_, 1024, 2048, nullptr, vraw, nullptr);

  // 4) per-head RMSNorm + RoPE, head-major relayout; v passthrough
  head_norm_rope_kern<<<(B_*T_*H_)/8, 256, 0, stream>>>(
      qraw, q_norm_w, cosT, sinT, qall, H_, T_, 0, 0);
  head_norm_rope_kern<<<(B_*T_*HKV_)/8, 256, 0, stream>>>(
      kraw, k_norm_w, cosT, sinT, kall, HKV_, TK_, R_, 1);
  head_norm_rope_kern<<<(B_*T_*HKV_)/8, 256, 0, stream>>>(
      vraw, k_norm_w, cosT, sinT, vall, HKV_, TK_, R_, 2);

  // 5) virtual-KV injection at token slots [0, R)
  inject_extras_kern<<<(B_*HKV_*R_*HD_)/256, 256, 0, stream>>>(
      key_extra, value_extra, alpha_logit, kall, vall);

  // 6) flash attention (WMMA)
  attn_kern<<<dim3(T_/128, H_, B_), 256, 0, stream>>>(qall, kall, vall, attn);

  // 7) Wo projection + residual -> hidden (f32)
  gemm_bf16_kern<<<dim3(2048/64, BT_/128), 256, 0, stream>>>(
      attn, wo_b, BT_, 2048, 2048, hid, nullptr, hidden_states);

  // 8) ln2
  rmsnorm_kern<<<BT_, 256, 0, stream>>>(hid, ln2_w, hn_b);

  // 9) MLP: gate/up -> SwiGLU -> down (+residual) into d_out
  gemm_bf16_kern<<<dim3(DFF_/64, BT_/128), 256, 0, stream>>>(
      hn_b, wg_b, BT_, DFF_, 2048, nullptr, g_b, nullptr);
  gemm_bf16_kern<<<dim3(DFF_/64, BT_/128), 256, 0, stream>>>(
      hn_b, wu_b, BT_, DFF_, 2048, nullptr, u_b, nullptr);
  swiglu_kern<<<8192, 256, 0, stream>>>(g_b, u_b, m_b, (size_t)BT_ * DFF_);
  gemm_bf16_kern<<<dim3(2048/64, BT_/128), 256, 0, stream>>>(
      m_b, wd_b, BT_, 2048, DFF_, (float*)d_out, nullptr, hid);
}
